// GraphTransformer_22333829939713
// MI455X (gfx1250) — matrix operand order repaired and checked by reading, not compiled, not run on hardware
//
#include <hip/hip_runtime.h>
#include <hip/hip_bf16.h>
#include <cmath>

// ---------------- problem constants (from reference) ----------------
#define NNODES 50000
#define NEDGES 800000

// ---------------- WMMA vector types ----------------
typedef __attribute__((ext_vector_type(16))) _Float16 v16h;
typedef __attribute__((ext_vector_type(8)))  _Float16 v8h;
typedef __attribute__((ext_vector_type(8)))  float    v8f;

union Frag16 { v16h v; v8h h[2]; };

// ---------------- helpers ----------------
__device__ __forceinline__ float waveReduceSum(float v) {
  #pragma unroll
  for (int off = 16; off >= 1; off >>= 1) v += __shfl_xor(v, off, 32);
  return v;
}
__device__ __forceinline__ float waveReduceMax(float v) {
  #pragma unroll
  for (int off = 16; off >= 1; off >>= 1) v = fmaxf(v, __shfl_xor(v, off, 32));
  return v;
}

// ---------------- f32 -> f16 convert ----------------
__global__ void cvt_f32_f16_kernel(const float* __restrict__ in,
                                   _Float16* __restrict__ out, int n) {
  int i = blockIdx.x * blockDim.x + threadIdx.x;
  if (i < n) out[i] = (_Float16)in[i];
}

// ---------------- fused Q/K/V/S projection via WMMA ----------------
// C[N, fo] = X[N, fi] @ W^T + b  for four weight sets sharing the A fragment.
// W is row-major [fo, fi], so B[k][n] = W[n][k]: per-lane B loads are
// contiguous rows of W. Layout used here (contiguous k order, same for A and B):
//   A (16x32 f16): lane L -> row M=L&15; K = (L>>4)*16 + {0..15}
//   B (32x16 f16): lane L -> col N=L&15; K = (L>>4)*16 + {0..15}
//   C (16x16 f32): lane L -> col N=L&15; vgpr i -> row M = i + 8*(L>>4)
__global__ __launch_bounds__(32)
void proj_wmma_kernel(const _Float16* __restrict__ X,
                      const _Float16* __restrict__ Wq, const _Float16* __restrict__ Wk,
                      const _Float16* __restrict__ Wv, const _Float16* __restrict__ Ws,
                      const float* __restrict__ bq, const float* __restrict__ bk,
                      const float* __restrict__ bv, const float* __restrict__ bs,
                      float* __restrict__ Q, float* __restrict__ K,
                      float* __restrict__ V, float* __restrict__ S,
                      int fi, int fo, int tilesN) {
  const int tile = blockIdx.x;
  const int tm = tile / tilesN;
  const int tn = tile % tilesN;
  const int lane = threadIdx.x;
  const int l15 = lane & 15;
  const int hi  = lane >> 4;

  const _Float16* xrow = X  + (size_t)(tm * 16 + l15) * fi;
  const _Float16* wqr  = Wq + (size_t)(tn * 16 + l15) * fi;
  const _Float16* wkr  = Wk + (size_t)(tn * 16 + l15) * fi;
  const _Float16* wvr  = Wv + (size_t)(tn * 16 + l15) * fi;
  const _Float16* wsr  = Ws + (size_t)(tn * 16 + l15) * fi;

  v8f cq = {}, ck = {}, cv = {}, cs = {};

  for (int kk = 0; kk < fi; kk += 32) {
    Frag16 A;
    A.h[0] = *(const v8h*)(xrow + kk + hi * 16);
    A.h[1] = *(const v8h*)(xrow + kk + hi * 16 + 8);

    Frag16 B;
    const int boff = kk + hi * 16;
    B.h[0] = *(const v8h*)(wqr + boff);
    B.h[1] = *(const v8h*)(wqr + boff + 8);
    cq = __builtin_amdgcn_wmma_f32_16x16x32_f16(false, A.v, false, B.v,
                                                (short)0, cq, false, false);
    B.h[0] = *(const v8h*)(wkr + boff);
    B.h[1] = *(const v8h*)(wkr + boff + 8);
    ck = __builtin_amdgcn_wmma_f32_16x16x32_f16(false, A.v, false, B.v,
                                                (short)0, ck, false, false);
    B.h[0] = *(const v8h*)(wvr + boff);
    B.h[1] = *(const v8h*)(wvr + boff + 8);
    cv = __builtin_amdgcn_wmma_f32_16x16x32_f16(false, A.v, false, B.v,
                                                (short)0, cv, false, false);
    B.h[0] = *(const v8h*)(wsr + boff);
    B.h[1] = *(const v8h*)(wsr + boff + 8);
    cs = __builtin_amdgcn_wmma_f32_16x16x32_f16(false, A.v, false, B.v,
                                                (short)0, cs, false, false);
  }

  const int colg = tn * 16 + l15;
  const float bqv = bq[colg], bkv = bk[colg], bvv = bv[colg], bsv = bs[colg];
  #pragma unroll
  for (int i = 0; i < 8; ++i) {
    const int row = tm * 16 + i + 8 * hi;
    const size_t idx = (size_t)row * fo + colg;
    Q[idx] = cq[i] + bqv;
    K[idx] = ck[i] + bkv;
    V[idx] = cv[i] + bvv;
    S[idx] = cs[i] + bsv;   // skip projection seeds the aggregation buffer
  }
}

// ---------------- per-node init: m = -inf, denom = 0 ----------------
__global__ void init_md_kernel(float* __restrict__ m, float* __restrict__ den, int n) {
  int i = blockIdx.x * blockDim.x + threadIdx.x;
  if (i < n) { m[i] = -INFINITY; den[i] = 0.0f; }
}

// ---------------- pass 1: edge score + segment atomic max ----------------
__global__ void score_kernel(const float* __restrict__ q, const float* __restrict__ k,
                             const long long* __restrict__ src,
                             const long long* __restrict__ dst,
                             float* __restrict__ score, float* __restrict__ m,
                             int E, int d, float inv_sqrt_d) {
  const int e = blockIdx.x * (blockDim.x >> 5) + (threadIdx.x >> 5);
  if (e >= E) return;
  const int lane = threadIdx.x & 31;
  const int sn = (int)src[e];
  const int tn = (int)dst[e];
  float p = 0.0f;
  for (int c = lane; c < d; c += 32)
    p += q[(size_t)tn * d + c] * k[(size_t)sn * d + c];
  p = waveReduceSum(p);
  if (lane == 0) {
    const float sc = p * inv_sqrt_d;
    score[e] = sc;
    // float atomic max via ordered-int trick (m init to -inf)
    if (sc >= 0.0f) atomicMax((int*)(m + tn), __float_as_int(sc));
    else            atomicMin((unsigned int*)(m + tn), __float_as_uint(sc));
  }
}

// ---------------- pass 2: exp(score - m[dst]) + segment sum ----------------
__global__ void expsum_kernel(const long long* __restrict__ dst,
                              float* __restrict__ score,
                              const float* __restrict__ m,
                              float* __restrict__ den, int E) {
  const int e = blockIdx.x * blockDim.x + threadIdx.x;
  if (e >= E) return;
  const int tn = (int)dst[e];
  const float ev = expf(score[e] - m[tn]);
  score[e] = ev;                 // overwrite in place, saves a scratch buffer
  atomicAdd(den + tn, ev);
}

// ---------------- pass 3: scatter alpha * v[src] into agg[dst] ----------------
__global__ void scatter_kernel(const float* __restrict__ v,
                               const long long* __restrict__ src,
                               const long long* __restrict__ dst,
                               const float* __restrict__ score,
                               const float* __restrict__ den,
                               float* __restrict__ agg, int E, int d) {
  const int e = blockIdx.x * (blockDim.x >> 5) + (threadIdx.x >> 5);
  if (e >= E) return;
  const int lane = threadIdx.x & 31;
  const int sn = (int)src[e];
  const int tn = (int)dst[e];
  const float alpha = score[e] / fmaxf(den[tn], 1e-16f);
  for (int c = lane; c < d; c += 32)
    atomicAdd(agg + (size_t)tn * d + c, alpha * v[(size_t)sn * d + c]);
}

// ---------------- in-place ReLU ----------------
__global__ void relu_kernel(float* __restrict__ x, int n) {
  int i = blockIdx.x * blockDim.x + threadIdx.x;
  if (i < n) x[i] = fmaxf(x[i], 0.0f);
}

// ---------------- log_softmax over d=32 (one wave per node) ----------------
__global__ void log_softmax32_kernel(const float* __restrict__ in,
                                     float* __restrict__ out, int n) {
  const int node = blockIdx.x * (blockDim.x >> 5) + (threadIdx.x >> 5);
  if (node >= n) return;
  const int lane = threadIdx.x & 31;
  const float v = in[(size_t)node * 32 + lane];
  const float mx = waveReduceMax(v);
  const float s = waveReduceSum(expf(v - mx));
  out[(size_t)node * 32 + lane] = v - mx - logf(s);
}

// ===================================================================
extern "C" void kernel_launch(void* const* d_in, const int* in_sizes, int n_in,
                              void* d_out, int out_size, void* d_ws, size_t ws_size,
                              hipStream_t stream) {
  (void)in_sizes; (void)n_in; (void)out_size; (void)ws_size;
  const int N = NNODES, E = NEDGES;

  const float* x = (const float*)d_in[0];
  const long long* ei = (const long long*)d_in[1];
  const long long* src = ei;       // edge_index[0, :]
  const long long* dst = ei + E;   // edge_index[1, :]

  // ---- workspace carve-out (256B aligned) ----
  char* ws = (char*)d_ws;
  size_t off = 0;
  auto take = [&](size_t bytes) -> char* {
    char* p = ws + off;
    off += (bytes + 255) & ~(size_t)255;
    return p;
  };
  _Float16* H16 = (_Float16*)take((size_t)N * 128 * sizeof(_Float16));
  float* Q   = (float*)take((size_t)N * 96 * sizeof(float));
  float* Kb  = (float*)take((size_t)N * 96 * sizeof(float));
  float* V   = (float*)take((size_t)N * 96 * sizeof(float));
  float* B0  = (float*)take((size_t)N * 96 * sizeof(float));  // agg / layer out (ping)
  float* B1  = (float*)take((size_t)N * 96 * sizeof(float));  // agg / layer out (pong)
  float* SC  = (float*)take((size_t)E * sizeof(float));       // score, then exp(score-m)
  float* M   = (float*)take((size_t)N * sizeof(float));
  float* DEN = (float*)take((size_t)N * sizeof(float));
  const size_t WSLOT = (size_t)128 * 96;
  _Float16* W16 = (_Float16*)take(4 * WSLOT * sizeof(_Float16));

  auto run_layer = [&](const float* hin, int fi, int fo, int wbase,
                       float* agg, bool do_relu) {
    // 1) convert activations + weights to f16
    const int nh = N * fi;
    cvt_f32_f16_kernel<<<(nh + 255) / 256, 256, 0, stream>>>(hin, H16, nh);
    const int nw = fo * fi;
    _Float16* w16[4];
    for (int i = 0; i < 4; ++i) {
      w16[i] = W16 + i * WSLOT;
      cvt_f32_f16_kernel<<<(nw + 255) / 256, 256, 0, stream>>>(
          (const float*)d_in[wbase + i], w16[i], nw);
    }
    const float* bq = (const float*)d_in[wbase + 4];
    const float* bk = (const float*)d_in[wbase + 5];
    const float* bv = (const float*)d_in[wbase + 6];
    const float* bs = (const float*)d_in[wbase + 7];

    // 2) fused WMMA projections (S written directly into agg)
    const int tilesM = N / 16, tilesN = fo / 16;
    proj_wmma_kernel<<<tilesM * tilesN, 32, 0, stream>>>(
        H16, w16[0], w16[1], w16[2], w16[3], bq, bk, bv, bs,
        Q, Kb, V, agg, fi, fo, tilesN);

    // 3) segment softmax attention over edges
    init_md_kernel<<<(N + 255) / 256, 256, 0, stream>>>(M, DEN, N);
    score_kernel<<<(E + 7) / 8, 256, 0, stream>>>(
        Q, Kb, src, dst, SC, M, E, fo, 1.0f / sqrtf((float)fo));
    expsum_kernel<<<(E + 255) / 256, 256, 0, stream>>>(dst, SC, M, DEN, E);
    scatter_kernel<<<(E + 7) / 8, 256, 0, stream>>>(V, src, dst, SC, DEN, agg, E, fo);

    // 4) activation
    if (do_relu) {
      const int no = N * fo;
      relu_kernel<<<(no + 255) / 256, 256, 0, stream>>>(agg, no);
    }
  };

  // layer 1: x[50000,128] -> B0[50000,96]
  run_layer(x, 128, 96, 2, B0, true);
  // layer 2: B0 -> B1[50000,96]
  run_layer(B0, 96, 96, 10, B1, true);
  // layer 3: B1 -> B0[50000,32] (no relu)
  run_layer(B1, 96, 32, 18, B0, false);

  // log_softmax over 32 channels -> d_out (f32)
  log_softmax32_kernel<<<(N + 7) / 8, 256, 0, stream>>>(B0, (float*)d_out, N);
}
